// HeteroSAGE_30605936951692
// MI455X (gfx1250) — matrix operand order repaired and checked by reading, not compiled, not run on hardware
//
#include <hip/hip_runtime.h>

// ---------------------------------------------------------------------------
// HeteroSAGE on MI455X (gfx1250, wave32).
// Strategy: project-then-aggregate (mean commutes with the linear map), so
// every scatter moves only H=32 floats per edge instead of 128/64. Dense
// projections run on v_wmma_f32_16x16x32_f16 (f32 accumulate). Scatters use
// native fp32 global atomics that stay resident in the 192MB L2.
// ---------------------------------------------------------------------------

typedef __attribute__((ext_vector_type(16))) _Float16 v16h;
typedef __attribute__((ext_vector_type(8)))  float    v8f;

#define NPAPER  200000
#define NAUTHOR 100000
#define NEDGE   1000000

// ---------------------------------------------------------------------------
// WMMA GEMM: O_j = A @ (Wa_j [+ Wb_j]) for j < NOUT.
//   A : N x K fp32 row-major (N multiple of 16, K in {32,64,128})
//   W : K x 32 fp32 row-major
//   O : N x 32 fp32 row-major
// One wave computes one 16-row tile (2 x 16x16 accumulators). Weights staged
// in LDS as f16, column-major with +2 halves padding (bank-conflict-free
// B-fragment builds). A fragments filled in contiguous k order (k = 16*half + s), same as B.
// ---------------------------------------------------------------------------
template <int K, int NOUT>
__global__ __launch_bounds__(256) void gemm32_wmma(
    const float* __restrict__ A, int N,
    const float* __restrict__ Wa0, const float* __restrict__ Wb0, float* __restrict__ O0,
    const float* __restrict__ Wa1, const float* __restrict__ Wb1, float* __restrict__ O1,
    const float* __restrict__ Wa2, const float* __restrict__ Wb2, float* __restrict__ O2)
{
    constexpr int KP     = K + 2;      // padded column stride (halves)
    constexpr int KSTEPS = K / 32;
    __shared__ _Float16 lw[NOUT * 32 * KP];

    const int tid = threadIdx.x;
    const float* Was[3] = {Wa0, Wa1, Wa2};
    const float* Wbs[3] = {Wb0, Wb1, Wb2};
    float*       Os [3] = {O0,  O1,  O2};

    // Stage (summed) weights into LDS, f16 column-major padded.
#pragma unroll
    for (int o = 0; o < NOUT; ++o) {
        const float* wa = Was[o];
        const float* wb = Wbs[o];
        for (int idx = tid; idx < K * 32; idx += 256) {
            const int k = idx >> 5, col = idx & 31;
            float w = wa[idx];
            if (wb) w += wb[idx];
            lw[o * 32 * KP + col * KP + k] = (_Float16)w;
        }
    }
    __syncthreads();

    const int lane  = tid & 31;
    const int wid   = tid >> 5;
    const int rlane = lane & 15;   // row within tile (A) / column within tile (B,D)
    const int half  = lane >> 4;
    const int tiles = N >> 4;

    for (int rt = blockIdx.x * 8 + wid; rt < tiles; rt += gridDim.x * 8) {
        const int row = rt * 16 + rlane;
        const float* ap = A + (size_t)row * K;

        // A fragments for all K-steps (kept in registers: loops fully unrolled).
        v16h afrag[KSTEPS];
#pragma unroll
        for (int kk = 0; kk < KSTEPS; ++kk) {
            const int base0 = kk * 32 + half * 16;  // K of frag elems 0..7
            const int base1 = base0 + 8;            // K of frag elems 8..15
            v16h a;
#pragma unroll
            for (int v = 0; v < 4; ++v) {
                const float2 f0 = *(const float2*)(ap + base0 + 2 * v);
                const float2 f1 = *(const float2*)(ap + base1 + 2 * v);
                a[2 * v]         = (_Float16)f0.x;
                a[2 * v + 1]     = (_Float16)f0.y;
                a[8 + 2 * v]     = (_Float16)f1.x;
                a[8 + 2 * v + 1] = (_Float16)f1.y;
            }
            afrag[kk] = a;
        }

#pragma unroll
        for (int o = 0; o < NOUT; ++o) {
            float* Op = Os[o];
            const _Float16* lwo = lw + o * 32 * KP;
#pragma unroll
            for (int nt = 0; nt < 2; ++nt) {
                const int col = nt * 16 + rlane;    // output column of this lane
                v8f acc = {};
#pragma unroll
                for (int kk = 0; kk < KSTEPS; ++kk) {
                    const int kbase = kk * 32 + half * 16;
                    const _Float16* wp = lwo + col * KP + kbase;
                    v16h b;
#pragma unroll
                    for (int e = 0; e < 16; ++e) b[e] = wp[e];
                    acc = __builtin_amdgcn_wmma_f32_16x16x32_f16(
                        false, afrag[kk], false, b, (short)0, acc, false, false);
                }
                // D layout: VGPR r -> row r + 8*half, column = rlane (+ nt*16)
#pragma unroll
                for (int r = 0; r < 8; ++r) {
                    const int m = rt * 16 + r + half * 8;
                    Op[(size_t)m * 32 + nt * 16 + rlane] = acc[r];
                }
            }
        }
    }
}

// ---------------------------------------------------------------------------
// Scatter-accumulate: msum[dst] += H[src] (32 dims / edge, one lane per dim).
// dst/src loads are wave-uniform; fp32 adds go to native L2 atomics.
// ---------------------------------------------------------------------------
__global__ __launch_bounds__(256) void scatter_accum(
    const float* __restrict__ H, const int* __restrict__ src,
    const int* __restrict__ dst, int E,
    float* __restrict__ msum, float* __restrict__ cnt)
{
    const long long t = (long long)blockIdx.x * blockDim.x + threadIdx.x;
    if (t >= (long long)E * 32) return;
    const int e = (int)(t >> 5);
    const int d = (int)(t & 31);
    const int s  = src[e];
    const int dd = dst[e];
    const float v = H[(size_t)s * 32 + d];
    __hip_atomic_fetch_add(&msum[(size_t)dd * 32 + d], v,
                           __ATOMIC_RELAXED, __HIP_MEMORY_SCOPE_AGENT);
    if (cnt && d == 0)
        __hip_atomic_fetch_add(&cnt[dd], 1.0f,
                               __ATOMIC_RELAXED, __HIP_MEMORY_SCOPE_AGENT);
}

// acc = relu(acc/max(cnt_c,1) + bc + accw/max(cnt_w,1) + bw + self)   (in place)
__global__ __launch_bounds__(256) void combine2_relu(
    float* __restrict__ acc_c, const float* __restrict__ acc_w,
    const float* __restrict__ selfp,
    const float* __restrict__ cnt_c, const float* __restrict__ cnt_w,
    const float* __restrict__ bc, const float* __restrict__ bw, int N)
{
    const int t = blockIdx.x * blockDim.x + threadIdx.x;
    if (t >= N * 32) return;
    const int i = t >> 5, d = t & 31;
    const float ic = 1.0f / fmaxf(cnt_c[i], 1.0f);
    const float iw = 1.0f / fmaxf(cnt_w[i], 1.0f);
    const float v = acc_c[t] * ic + bc[d] + acc_w[t] * iw + bw[d] + selfp[t];
    acc_c[t] = fmaxf(v, 0.0f);
}

// acc = relu(acc/max(cnt,1) + bias + self)   (in place)
__global__ __launch_bounds__(256) void combine1_relu(
    float* __restrict__ acc, const float* __restrict__ selfp,
    const float* __restrict__ cnt, const float* __restrict__ bias, int N)
{
    const int t = blockIdx.x * blockDim.x + threadIdx.x;
    if (t >= N * 32) return;
    const int i = t >> 5, d = t & 31;
    const float ic = 1.0f / fmaxf(cnt[i], 1.0f);
    const float v = acc[t] * ic + bias[d] + selfp[t];
    acc[t] = fmaxf(v, 0.0f);
}

// out[i] = dot( acc_c/max(c0,1)+bc + acc_w/max(c1,1)+bw + self , Wh ) + bh
// One wave per node; __shfl_xor tree reduction across the 32 lanes.
__global__ __launch_bounds__(256) void head_kernel(
    const float* __restrict__ acc_c, const float* __restrict__ acc_w,
    const float* __restrict__ selfp,
    const float* __restrict__ cnt_c, const float* __restrict__ cnt_w,
    const float* __restrict__ bc, const float* __restrict__ bw,
    const float* __restrict__ Wh, const float* __restrict__ bh,
    float* __restrict__ out, int N)
{
    const int t = blockIdx.x * blockDim.x + threadIdx.x;
    const int i = t >> 5, d = t & 31;
    if (i >= N) return;
    const float ic = 1.0f / fmaxf(cnt_c[i], 1.0f);
    const float iw = 1.0f / fmaxf(cnt_w[i], 1.0f);
    const size_t idx = (size_t)i * 32 + d;
    const float p2 = acc_c[idx] * ic + bc[d] + acc_w[idx] * iw + bw[d] + selfp[idx];
    float v = p2 * Wh[d];
#pragma unroll
    for (int off = 16; off > 0; off >>= 1)
        v += __shfl_xor(v, off, 32);
    if (d == 0) out[i] = v + bh[0];
}

// ---------------------------------------------------------------------------
extern "C" void kernel_launch(void* const* d_in, const int* in_sizes, int n_in,
                              void* d_out, int out_size, void* d_ws, size_t ws_size,
                              hipStream_t stream)
{
    (void)in_sizes; (void)n_in; (void)out_size; (void)ws_size;

    const float* x_paper    = (const float*)d_in[0];
    const float* x_author   = (const float*)d_in[1];
    const int*   ei_cites   = (const int*)d_in[2];
    const int*   ei_writes  = (const int*)d_in[3];
    const int*   ei_rev     = (const int*)d_in[4];
    const float* W1l_cites  = (const float*)d_in[5];
    const float* b1l_cites  = (const float*)d_in[6];
    const float* W1r_cites  = (const float*)d_in[7];
    const float* W1l_writes = (const float*)d_in[8];
    const float* b1l_writes = (const float*)d_in[9];
    const float* W1r_writes = (const float*)d_in[10];
    const float* W1l_rev    = (const float*)d_in[11];
    const float* b1l_rev    = (const float*)d_in[12];
    const float* W1r_rev    = (const float*)d_in[13];
    const float* W2l_cites  = (const float*)d_in[14];
    const float* b2l_cites  = (const float*)d_in[15];
    const float* W2r_cites  = (const float*)d_in[16];
    const float* W2l_writes = (const float*)d_in[17];
    const float* b2l_writes = (const float*)d_in[18];
    const float* W2r_writes = (const float*)d_in[19];
    // d_in[20..22] (W2*_rev): dead — the author branch of layer 2 is unused.
    const float* Wh = (const float*)d_in[23];
    const float* bh = (const float*)d_in[24];
    float* out = (float*)d_out;

    const size_t NP32 = (size_t)NPAPER * 32;
    const size_t NA32 = (size_t)NAUTHOR * 32;
    float* ws = (float*)d_ws;
    // Buffer plan (aliased across phases; ~169 MB total):
    float* A0 = ws;            // hp_cites      -> layer2 msum_cites
    float* A1 = A0 + NP32;     // hp_rev        -> layer2 hp2 (p1 @ W2l_cites)
    float* A2 = A1 + NP32;     // paper self(1) -> paper self(2)
    float* A3 = A2 + NP32;     // msum_cites    -> p1 (in place)
    float* A4 = A3 + NP32;     // msum_writes   -> layer2 msum_writes
    float* B0 = A4 + NP32;     // ha_writes     -> layer2 ha2 (a1 @ W2l_writes)
    float* B1 = B0 + NA32;     // author self(1)
    float* B2 = B1 + NA32;     // msum_rev      -> a1 (in place)
    float* C0 = B2 + NA32;     // cnt_cites  [NPAPER]  (reused by layer 2)
    float* C1 = C0 + NPAPER;   // cnt_writes [NPAPER]  (reused by layer 2)
    float* C2 = C1 + NPAPER;   // cnt_rev    [NAUTHOR]

    // Zero layer-1 accumulators + counts (A3,A4 contiguous; B2,C0,C1,C2 contiguous).
    hipMemsetAsync(A3, 0, 2 * NP32 * sizeof(float), stream);
    hipMemsetAsync(B2, 0, (NA32 + 2 * (size_t)NPAPER + NAUTHOR) * sizeof(float), stream);

    // ---- Layer 1 projections (x read once; fused multi-output GEMMs) ----
    gemm32_wmma<128, 3><<<(NPAPER / 16 + 7) / 8, 256, 0, stream>>>(
        x_paper, NPAPER,
        W1l_cites, nullptr,    A0,
        W1l_rev,   nullptr,    A1,
        W1r_cites, W1r_writes, A2);
    gemm32_wmma<64, 2><<<(NAUTHOR / 16 + 7) / 8, 256, 0, stream>>>(
        x_author, NAUTHOR,
        W1l_writes, nullptr, B0,
        W1r_rev,    nullptr, B1,
        nullptr, nullptr, nullptr);

    // ---- Layer 1 scatters (payload already reduced to 32 dims) ----
    const int sblocks = (int)(((size_t)NEDGE * 32 + 255) / 256);
    scatter_accum<<<sblocks, 256, 0, stream>>>(A0, ei_cites,  ei_cites  + NEDGE, NEDGE, A3, C0);
    scatter_accum<<<sblocks, 256, 0, stream>>>(B0, ei_writes, ei_writes + NEDGE, NEDGE, A4, C1);
    scatter_accum<<<sblocks, 256, 0, stream>>>(A1, ei_rev,    ei_rev    + NEDGE, NEDGE, B2, C2);

    const int cpb = (int)((NP32 + 255) / 256);
    const int cab = (int)((NA32 + 255) / 256);
    combine2_relu<<<cpb, 256, 0, stream>>>(A3, A4, A2, C0, C1, b1l_cites, b1l_writes, NPAPER);
    combine1_relu<<<cab, 256, 0, stream>>>(B2, B1, C2, b1l_rev, NAUTHOR);
    // p1 = A3, a1 = B2.

    // ---- Layer 2 ----
    hipMemsetAsync(A0, 0, NP32 * sizeof(float), stream);
    hipMemsetAsync(A4, 0, NP32 * sizeof(float), stream);

    gemm32_wmma<32, 2><<<(NPAPER / 16 + 7) / 8, 256, 0, stream>>>(
        A3, NPAPER,
        W2l_cites, nullptr,    A1,
        W2r_cites, W2r_writes, A2,
        nullptr, nullptr, nullptr);
    gemm32_wmma<32, 1><<<(NAUTHOR / 16 + 7) / 8, 256, 0, stream>>>(
        B2, NAUTHOR,
        W2l_writes, nullptr, B0,
        nullptr, nullptr, nullptr,
        nullptr, nullptr, nullptr);

    // Degree counts are edge-only: reuse C0/C1, skip count atomics.
    scatter_accum<<<sblocks, 256, 0, stream>>>(A1, ei_cites,  ei_cites  + NEDGE, NEDGE, A0, nullptr);
    scatter_accum<<<sblocks, 256, 0, stream>>>(B0, ei_writes, ei_writes + NEDGE, NEDGE, A4, nullptr);

    // ---- Fused combine + 32->1 head ----
    head_kernel<<<cpb, 256, 0, stream>>>(A0, A4, A2, C0, C1,
                                         b2l_cites, b2l_writes, Wh, bh, out, NPAPER);
}
